// DPENet_NOADD_73924977098860
// MI455X (gfx1250) — matrix operand order repaired and checked by reading, not compiled, not run on hardware
//
#include <hip/hip_runtime.h>
#include <math.h>

// ---------------------------------------------------------------------------
// MI455X (gfx1250) DPENet: conv towers as implicit-GEMM on v_wmma_f32_16x16x32_bf16
// ---------------------------------------------------------------------------

typedef __attribute__((ext_vector_type(16))) __bf16 v16bf;
typedef __attribute__((ext_vector_type(8)))  __bf16 v8bf;
typedef __attribute__((ext_vector_type(8)))  float  v8f;

static constexpr int BATCH = 4;
static constexpr int HH = 256;
static constexpr int WW = 256;
static constexpr int HP = HH + 2;   // halo-padded
static constexpr int WP = WW + 2;
static constexpr int CINC = 32;     // every conv layer has Cin = 32
static constexpr int NC = 10;       // inputs channels

__device__ __forceinline__ float gelu_exact(float x) {
    return 0.5f * x * (1.0f + erff(x * 0.70710678118654752f));
}

// ---------------------------------------------------------------------------
// Async (ASYNCcnt-tracked) weight staging: global -> LDS, 16B per lane per op.
// ---------------------------------------------------------------------------
__device__ __forceinline__ void stage_weights_async(const __bf16* __restrict__ wpack,
                                                    __bf16* ldsw, int n16) {
    // low 32 bits of a flat shared-aperture address == wave-relative LDS offset
    unsigned dst = (unsigned)(uintptr_t)ldsw + (unsigned)threadIdx.x * 16u;
    unsigned long long src = (unsigned long long)(uintptr_t)wpack +
                             (unsigned long long)threadIdx.x * 16ull;
    for (int i = threadIdx.x; i < n16; i += 256) {
        asm volatile("global_load_async_to_lds_b128 %0, %1, off"
                     :: "v"(dst), "v"(src) : "memory");
        dst += 256u * 16u;
        src += 256ull * 16ull;
    }
    asm volatile("s_wait_asynccnt 0x0" ::: "memory");
}

// ---------------------------------------------------------------------------
// zero fill (uint4, grid-stride)
// ---------------------------------------------------------------------------
__global__ void zero_ws(uint4* p, size_t n16) {
    size_t i = (size_t)blockIdx.x * blockDim.x + threadIdx.x;
    size_t stride = (size_t)gridDim.x * blockDim.x;
    uint4 z; z.x = z.y = z.z = z.w = 0u;
    for (; i < n16; i += stride) p[i] = z;
}

// ---------------------------------------------------------------------------
// feat NCHW f32 -> padded NHWC bf16
// ---------------------------------------------------------------------------
__global__ void pack_feat(const float* __restrict__ feat, __bf16* __restrict__ featP) {
    size_t idx = (size_t)blockIdx.x * blockDim.x + threadIdx.x;
    size_t total = (size_t)BATCH * HH * WW * CINC;
    if (idx >= total) return;
    int c = idx & 31; size_t r = idx >> 5;
    int x = (int)(r & (WW - 1)); r >>= 8;
    int y = (int)(r & (HH - 1)); int b = (int)(r >> 8);
    featP[((size_t)(b * HP + (y + 1)) * WP + (x + 1)) * CINC + c] =
        (__bf16)feat[(((size_t)(b * CINC + c) * HH + y) * WW + x)];
}

// ---------------------------------------------------------------------------
// weights (Cout,32,3,3) f32 -> B-operand-packed bf16, Cout padded (zeros).
// Per (tap,nt) block of 512 bf16:
//   [part(k>>3)][lane(=h*16+n)][k&7]  -> each ds_load_b128 has 16B lane stride
//   (conflict-free: 32 lanes cover 512B contiguous).
// Lane L(=half*16+col) reads K{8h..8h+7} at  block*512 + L*8
//                  and K{16+8h..16+8h+7} -> lo = part0, hi = part1 (+256 elements).
// ---------------------------------------------------------------------------
__global__ void pack_w(const float* __restrict__ w, __bf16* __restrict__ out,
                       int Cout, int CoutPad) {
    int idx = blockIdx.x * blockDim.x + threadIdx.x;
    int total = 9 * CoutPad * CINC;
    if (idx >= total) return;
    int k  = idx & 15;
    int n  = (idx >> 4) & 15;
    int h  = (idx >> 8) & 1;
    int t2 = idx >> 9;            // nt*9 + tap
    int tap = t2 % 9;
    int nt  = t2 / 9;
    int cin  = (k >> 3) * 16 + h * 8 + (k & 7);   // K index seen by lane half h
    int cout = nt * 16 + n;
    float v = (cout < Cout) ? w[((size_t)cout * CINC + cin) * 9 + tap] : 0.0f;
    out[(size_t)(tap * (CoutPad / 16) + nt) * 512 +
        (k >> 3) * 256 + (h * 16 + n) * 8 + (k & 7)] = (__bf16)v;
}

// ---------------------------------------------------------------------------
// Implicit-GEMM conv3x3 (Cin=32) via v_wmma_f32_16x16x32_bf16.
// Block = 256 threads = 8 waves; wave owns MT*16 pixels of row y.
// Weights staged in LDS (async engine) in B-packed conflict-free layout.
// MODE 0: +bias, exact GELU, store to padded NHWC bf16 (Cout==32).
// MODE 1: +bias, raw, store to flat NHWC bf16 (stride NP*32).
// ---------------------------------------------------------------------------
template <int NP, int MT, int MODE>
__global__ __launch_bounds__(256)
void conv_wmma(const __bf16* __restrict__ in, const __bf16* __restrict__ wpack,
               const float* __restrict__ bias, __bf16* __restrict__ out, int Cout) {
    constexpr int NT2 = NP * 2;                 // #16-wide N tiles
    extern __shared__ __bf16 ldsw[];
    stage_weights_async(wpack, ldsw, 9 * NT2 * 64);
    __syncthreads();

    const int wave   = threadIdx.x >> 5;
    const int lane   = threadIdx.x & 31;
    const int half   = lane >> 4;               // A/B lane half (K split)
    const int lanelo = lane & 15;               // A: row M, B: col N
    const int y  = blockIdx.y;
    const int b  = blockIdx.z;
    const int x0 = blockIdx.x * (128 * MT) + wave * (16 * MT);

    const v8f vz = {0.f, 0.f, 0.f, 0.f, 0.f, 0.f, 0.f, 0.f};
    v8f acc[NT2 * MT];
#pragma unroll
    for (int i = 0; i < NT2 * MT; ++i) acc[i] = vz;

    union AV { v16bf v; struct { v8bf lo, hi; } p; };

#pragma unroll
    for (int tap = 0; tap < 9; ++tap) {
        const int dy = tap / 3 - 1, dx = tap % 3 - 1;
        AV a[MT];
#pragma unroll
        for (int mt = 0; mt < MT; ++mt) {
            // A tile: lane half h holds K{8h..8h+7, 16+8h..16+8h+7} of its pixel
            const __bf16* arow = in +
                ((size_t)((b * HP + (y + 1 + dy)) * WP +
                          (x0 + mt * 16 + lanelo + 1 + dx))) * CINC + half * 8;
            a[mt].p.lo = *(const v8bf*)(arow);
            a[mt].p.hi = *(const v8bf*)(arow + 16);
        }
#pragma unroll
        for (int nt = 0; nt < NT2; ++nt) {
            const __bf16* bp = ldsw + (size_t)(tap * NT2 + nt) * 512 + lane * 8;
            AV bm;
            bm.p.lo = *(const v8bf*)(bp);          // part 0 (K 0..7 of this half)
            bm.p.hi = *(const v8bf*)(bp + 256);    // part 1 (K 8..15 of this half)
#pragma unroll
            for (int mt = 0; mt < MT; ++mt)
                acc[nt * MT + mt] = __builtin_amdgcn_wmma_f32_16x16x32_bf16(
                    false, a[mt].v, false, bm.v, (short)0, acc[nt * MT + mt],
                    false, false);
        }
    }

#pragma unroll
    for (int nt = 0; nt < NT2; ++nt) {
        const int ch = nt * 16 + lanelo;
        const float bb = (ch < Cout) ? bias[ch] : 0.0f;
#pragma unroll
        for (int mt = 0; mt < MT; ++mt) {
#pragma unroll
            for (int r = 0; r < 8; ++r) {
                const int m = mt * 16 + half * 8 + r;   // C/D: M = 8*(lane/16)+vgpr
                float v = acc[nt * MT + mt][r] + bb;
                if (MODE == 0) {
                    v = gelu_exact(v);
                    out[((size_t)((b * HP + (y + 1)) * WP + (x0 + m + 1))) * CINC + ch] =
                        (__bf16)v;
                } else {
                    out[((size_t)((b * HH + y) * WW + (x0 + m))) * (NT2 * 16) + ch] =
                        (__bf16)v;
                }
            }
        }
    }
}

// ---------------------------------------------------------------------------
// K-tower head: same WMMA conv, but reduce over pixels into per-block partial
// sums (deterministic: no float atomics).  CoutPad = 96 (NT2 = 6).
// ---------------------------------------------------------------------------
__global__ __launch_bounds__(256)
void conv_kmean(const __bf16* __restrict__ in, const __bf16* __restrict__ wpack,
                const float* __restrict__ bias, float* __restrict__ partial, int Cout) {
    constexpr int NT2 = 6;
    extern __shared__ __bf16 ldsw[];
    __shared__ float sred[8 * 96];
    stage_weights_async(wpack, ldsw, 9 * NT2 * 64);
    __syncthreads();

    const int wave   = threadIdx.x >> 5;
    const int lane   = threadIdx.x & 31;
    const int half   = lane >> 4;
    const int lanelo = lane & 15;
    const int y  = blockIdx.y;
    const int b  = blockIdx.z;
    const int x0 = blockIdx.x * 128 + wave * 16;

    const v8f vz = {0.f, 0.f, 0.f, 0.f, 0.f, 0.f, 0.f, 0.f};
    v8f acc[NT2];
#pragma unroll
    for (int i = 0; i < NT2; ++i) acc[i] = vz;

    union AV { v16bf v; struct { v8bf lo, hi; } p; };

#pragma unroll
    for (int tap = 0; tap < 9; ++tap) {
        const int dy = tap / 3 - 1, dx = tap % 3 - 1;
        const __bf16* arow = in +
            ((size_t)((b * HP + (y + 1 + dy)) * WP + (x0 + lanelo + 1 + dx))) * CINC + half * 8;
        AV a;
        a.p.lo = *(const v8bf*)(arow);
        a.p.hi = *(const v8bf*)(arow + 16);
#pragma unroll
        for (int nt = 0; nt < NT2; ++nt) {
            const __bf16* bp = ldsw + (size_t)(tap * NT2 + nt) * 512 + lane * 8;
            AV bm;
            bm.p.lo = *(const v8bf*)(bp);
            bm.p.hi = *(const v8bf*)(bp + 256);
            acc[nt] = __builtin_amdgcn_wmma_f32_16x16x32_bf16(
                false, a.v, false, bm.v, (short)0, acc[nt], false, false);
        }
    }

    // reduce the 16 pixels of this wave per channel
#pragma unroll
    for (int nt = 0; nt < NT2; ++nt) {
        const int ch = nt * 16 + lanelo;
        const float bb = (ch < Cout) ? bias[ch] : 0.0f;
        float s = 0.0f;
#pragma unroll
        for (int r = 0; r < 8; ++r) s += acc[nt][r] + bb;   // bias per pixel
        s += __shfl_xor(s, 16, 32);                          // fold lane <-> lane+16
        if (half == 0) sred[wave * 96 + ch] = s;
    }
    __syncthreads();
    const int t = threadIdx.x;
    if (t < 96) {
        float s = 0.0f;
#pragma unroll
        for (int wv = 0; wv < 8; ++wv) s += sred[wv * 96 + t];
        const int blin = (blockIdx.z * gridDim.y + blockIdx.y) * gridDim.x + blockIdx.x;
        partial[(size_t)blin * 96 + t] = s;
    }
}

__global__ void kreduce(const float* __restrict__ partial, float* __restrict__ kw,
                        int nblocks) {
    int t = threadIdx.x;
    if (t >= 96) return;
    float s = 0.0f;
    for (int i = 0; i < nblocks; ++i) s += partial[(size_t)i * 96 + t];
    kw[t] = s * (1.0f / ((float)BATCH * HH * WW));
}

// ---------------------------------------------------------------------------
// Deformable bilinear sampling + mask + per-channel kernel bank + bias.
// ---------------------------------------------------------------------------
__global__ __launch_bounds__(256)
void deform_sample(const float* __restrict__ inp, const __bf16* __restrict__ offb,
                   const __bf16* __restrict__ maskb, const float* __restrict__ kw,
                   const float* __restrict__ bias10, float* __restrict__ out) {
    const int x  = blockIdx.x * blockDim.x + threadIdx.x;
    const int y  = blockIdx.y;
    const int bc = blockIdx.z;
    const int b = bc / NC, c = bc % NC;
    if (x >= WW) return;

    const size_t pix = (size_t)(b * HH + y) * WW + x;
    const __bf16* op = offb + pix * 192;
    const __bf16* mp = maskb + pix * 96;
    const float*  ip = inp + (size_t)(b * NC + c) * HH * WW;

    float acc = 0.0f;
#pragma unroll
    for (int tap = 0; tap < 9; ++tap) {
        const float oy = (float)op[tap * 20 + c];        // (tap,0,c)
        const float ox = (float)op[tap * 20 + 10 + c];   // (tap,1,c)
        const float mr = (float)mp[tap * 10 + c];
        const float m  = 1.0f / (1.0f + expf(-mr));

        const float py = (float)y + (float)(tap / 3 - 1) + oy;
        const float px = (float)x + (float)(tap % 3 - 1) + ox;
        const float y0f = floorf(py), x0f = floorf(px);
        const float wy = py - y0f, wx = px - x0f;
        const int y0 = (int)y0f, xi0 = (int)x0f;

        auto g = [&](int yy, int xx) -> float {
            if (yy < 0 || yy >= HH || xx < 0 || xx >= WW) return 0.0f;
            return ip[(size_t)yy * WW + xx];
        };
        const float v00 = g(y0, xi0),     v01 = g(y0, xi0 + 1);
        const float v10 = g(y0 + 1, xi0), v11 = g(y0 + 1, xi0 + 1);
        const float s = v00 * (1.0f - wy) * (1.0f - wx) + v01 * (1.0f - wy) * wx +
                        v10 * wy * (1.0f - wx)          + v11 * wy * wx;
        acc += m * s * kw[c * 9 + tap];                  // wflat[c][tap] = k[c*9+tap]
    }
    out[(size_t)(b * NC + c) * HH * WW + (size_t)y * WW + x] = acc + bias10[c];
}

// ---------------------------------------------------------------------------
// host launcher
// ---------------------------------------------------------------------------
extern "C" void kernel_launch(void* const* d_in, const int* in_sizes, int n_in,
                              void* d_out, int out_size, void* d_ws, size_t ws_size,
                              hipStream_t stream) {
    (void)in_sizes; (void)n_in; (void)out_size; (void)ws_size;
    const float* feat   = (const float*)d_in[0];
    const float* inputs = (const float*)d_in[1];
    const float* tw[3][4]; const float* tb[3][4];
    for (int t = 0; t < 3; ++t)
        for (int j = 0; j < 4; ++j) {
            tw[t][j] = (const float*)d_in[2 + t * 8 + j * 2];
            tb[t][j] = (const float*)d_in[2 + t * 8 + j * 2 + 1];
        }
    const float* bias10 = (const float*)d_in[26];

    // ---- workspace carve-up (256B aligned) ----
    char* ws = (char*)d_ws;
    size_t off = 0;
    auto carve = [&](size_t bytes) -> char* {
        off = (off + 255) & ~(size_t)255;
        char* p = ws + off;
        off += bytes;
        return p;
    };
    const size_t actBytes = (size_t)BATCH * HP * WP * CINC * 2;   // padded NHWC bf16
    __bf16* featP = (__bf16*)carve(actBytes);
    __bf16* t1    = (__bf16*)carve(actBytes);
    __bf16* t2    = (__bf16*)carve(actBytes);
    __bf16* offb  = (__bf16*)carve((size_t)BATCH * HH * WW * 192 * 2);
    __bf16* maskb = (__bf16*)carve((size_t)BATCH * HH * WW * 96 * 2);
    const int convBlocks = (WW / 128) * HH * BATCH;               // 2048 (kmean grid)
    float* kpart = (float*)carve((size_t)convBlocks * 96 * 4);
    float* kwbuf = (float*)carve(96 * 4);
    auto wpBytes = [](int coutPad) { return (size_t)9 * coutPad * CINC * 2; };
    __bf16* wp[3][4];
    const int coutPad4[3] = {192, 96, 96};
    const int cout4[3]    = {180, 90, 90};
    for (int t = 0; t < 3; ++t) {
        for (int j = 0; j < 3; ++j) wp[t][j] = (__bf16*)carve(wpBytes(32));
        wp[t][3] = (__bf16*)carve(wpBytes(coutPad4[t]));
    }

    // ---- prep ----
    zero_ws<<<4096, 256, 0, stream>>>((uint4*)featP, 3 * actBytes / 16); // featP,t1,t2 contiguous
    {
        size_t total = (size_t)BATCH * HH * WW * CINC;
        pack_feat<<<(unsigned)((total + 255) / 256), 256, 0, stream>>>(feat, featP);
    }
    for (int t = 0; t < 3; ++t) {
        for (int j = 0; j < 3; ++j) {
            int tot = 9 * 32 * CINC;
            pack_w<<<(tot + 255) / 256, 256, 0, stream>>>(tw[t][j], wp[t][j], 32, 32);
        }
        int tot = 9 * coutPad4[t] * CINC;
        pack_w<<<(tot + 255) / 256, 256, 0, stream>>>(tw[t][3], wp[t][3], cout4[t], coutPad4[t]);
    }

    // ---- towers ----
    const dim3 gconvM2(WW / 256, HH, BATCH);   // MT=2: wave owns 32 px, block owns 256
    const dim3 gconvM1(WW / 128, HH, BATCH);   // MT=1
    const size_t sh32  = wpBytes(32);
    const size_t sh192 = wpBytes(192);
    const size_t sh96  = wpBytes(96);
    for (int t = 0; t < 3; ++t) {
        conv_wmma<1, 2, 0><<<gconvM2, 256, sh32, stream>>>(featP, wp[t][0], tb[t][0], t1, 32);
        conv_wmma<1, 2, 0><<<gconvM2, 256, sh32, stream>>>(t1,    wp[t][1], tb[t][1], t2, 32);
        conv_wmma<1, 2, 0><<<gconvM2, 256, sh32, stream>>>(t2,    wp[t][2], tb[t][2], t1, 32);
        if (t == 0)
            conv_wmma<6, 1, 1><<<gconvM1, 256, sh192, stream>>>(t1, wp[t][3], tb[t][3], offb, 180);
        else if (t == 1)
            conv_wmma<3, 1, 1><<<gconvM1, 256, sh96, stream>>>(t1, wp[t][3], tb[t][3], maskb, 90);
        else
            conv_kmean<<<gconvM1, 256, sh96, stream>>>(t1, wp[t][3], tb[t][3], kpart, 90);
    }
    kreduce<<<1, 96, 0, stream>>>(kpart, kwbuf, convBlocks);

    // ---- deformable sampling ----
    dim3 g2((WW + 255) / 256, HH, BATCH * NC);
    deform_sample<<<g2, 256, 0, stream>>>(inputs, offb, maskb, kwbuf, bias10,
                                          (float*)d_out);
}
